// Recurrence_46969762349501
// MI455X (gfx1250) — matrix operand order repaired and checked by reading, not compiled, not run on hardware
//
#include <hip/hip_runtime.h>
#include <stdint.h>

#define T_STEPS 32
#define NB      32
#define DOBS    64
#define HID     512
#define HEADS   4
#define MEMN    128
#define NACT    16
#define G3      1536
#define XI_RAW  3607
#define XI_PAD  3616

// xi section offsets (cumsum of SECTIONS)
#define OFF_KR 0
#define OFF_BR 2048
#define OFF_KW 2052
#define OFF_BW 2564
#define OFF_E  2565
#define OFF_V  3077
#define OFF_F  3589
#define OFF_GA 3593
#define OFF_GW 3594
#define OFF_PI 3595

typedef __attribute__((ext_vector_type(16))) _Float16 v16h;
typedef __attribute__((ext_vector_type(8)))  _Float16 v8h;
typedef __attribute__((ext_vector_type(8)))  float    v8f;

__device__ __forceinline__ v8f zero_v8f() {
    v8f z;
#pragma unroll
    for (int i = 0; i < 8; ++i) z[i] = 0.f;
    return z;
}

// A fragment (16x32 f16): row = lane&15; lanes<16 hold K 0..7 & 16..23, lanes>=16 hold 8..15 & 24..31
__device__ __forceinline__ v16h load_a_frag(const _Float16* A, int ld, int row0, int k0, int lane) {
    const _Float16* p = A + (size_t)(row0 + (lane & 15)) * ld + k0 + ((lane >> 4) << 3);
    v8h lo = *(const v8h*)p;
    v8h hi = *(const v8h*)(p + 16);
    v16h a;
#pragma unroll
    for (int i = 0; i < 8; ++i) { a[i] = lo[i]; a[i + 8] = hi[i]; }
    return a;
}

// B fragment (32x16 f16): col = lane&15; lanes<16 hold K 0..7 & 16..23, lanes>=16 hold 8..15 & 24..31
// weight stored row-major [out_col][K] so these are two contiguous 16B loads.
__device__ __forceinline__ v16h load_b_frag(const _Float16* B, int ld, int col0, int k0, int lane) {
    const _Float16* p = B + (size_t)(col0 + (lane & 15)) * ld + k0 + ((lane >> 4) << 3);
    v8h lo = *(const v8h*)p;
    v8h hi = *(const v8h*)(p + 16);
    v16h b;
#pragma unroll
    for (int i = 0; i < 8; ++i) { b[i] = lo[i]; b[i + 8] = hi[i]; }
    return b;
}

__device__ __forceinline__ v8f wmma16(v16h a, v16h b, v8f c) {
    return __builtin_amdgcn_wmma_f32_16x16x32_f16(false, a, false, b, (short)0, c, false, false);
}

__device__ __forceinline__ float wave_sum(float v) {
#pragma unroll
    for (int off = 16; off; off >>= 1) v += __shfl_xor(v, off, 32);
    return v;
}

__device__ __forceinline__ float sigm(float x) { return 1.f / (1.f + expf(-x)); }
__device__ __forceinline__ float sftp(float x) { return (x > 20.f) ? x : log1pf(expf(x)); }

// ---------------- pack / convert kernels ----------------
__global__ void k_cvt_f16(const float* __restrict__ src, _Float16* __restrict__ dst, int n) {
    int i = blockIdx.x * 256 + threadIdx.x;
    if (i < n) dst[i] = (_Float16)src[i];
}

__global__ void k_pack_wf(const float* __restrict__ Wf, _Float16* __restrict__ dst) {
    int i = blockIdx.x * 256 + threadIdx.x;   // over XI_PAD*HID
    if (i < XI_PAD * HID) {
        int r = i >> 9;
        dst[i] = (r < XI_RAW) ? (_Float16)Wf[(size_t)r * HID + (i & 511)] : (_Float16)0.f;
    }
}

// ---------------- GI = x @ w_ih.T + b_ih : [1024,64]x[64,1536] ----------------
__global__ void k_gi_gemm(const _Float16* __restrict__ x16, const _Float16* __restrict__ wih16,
                          const float* __restrict__ b_ih, float* __restrict__ GI) {
    int lane = threadIdx.x & 31, wave = threadIdx.x >> 5;
    int gid = blockIdx.x * 8 + wave;      // 6144 waves = 64 row-tiles x 96 col-tiles
    int rt = gid & 63;
    int ct = gid >> 6;
    v8f acc = zero_v8f();
#pragma unroll
    for (int kc = 0; kc < 2; ++kc) {
        v16h a = load_a_frag(x16, DOBS, rt * 16, kc * 32, lane);
        v16h b = load_b_frag(wih16, DOBS, ct * 16, kc * 32, lane);
        acc = wmma16(a, b, acc);
    }
    int col = ct * 16 + (lane & 15);
    int r0  = rt * 16 + 8 * (lane >> 4);
    float bias = b_ih[col];
#pragma unroll
    for (int g = 0; g < 8; ++g) GI[(size_t)(r0 + g) * G3 + col] = acc[g] + bias;
}

// ---------------- GRU recurrence (single persistent WG, 16 waves) ----------------
__global__ void k_gru(const float* __restrict__ GI, const _Float16* __restrict__ whh16,
                      const float* __restrict__ b_hh, const float* __restrict__ Wc,
                      const float* __restrict__ bc, _Float16* __restrict__ Hrelu16,
                      float* __restrict__ out) {
    extern __shared__ float smem[];
    float*    h32   = smem;                         // 32*512 f32
    float*    stage = h32 + NB * HID;               // 32*1536 f32
    _Float16* hh16  = (_Float16*)(stage + NB * G3); // 32*512 f16

    const int tid = threadIdx.x, lane = tid & 31, wave = tid >> 5;

    for (int i = tid; i < NB * HID; i += 512) { h32[i] = 0.f; hh16[i] = (_Float16)0.f; }
    __syncthreads();

    for (int t = 0; t < T_STEPS; ++t) {
        // gh = h @ w_hh.T  : [32,512]x[512,1536]; wave handles 6 col-tiles, 2 row-tiles
        v8f acc[2][6];
#pragma unroll
        for (int r = 0; r < 2; ++r)
#pragma unroll
            for (int c = 0; c < 6; ++c) acc[r][c] = zero_v8f();

        for (int kc = 0; kc < 16; ++kc) {
            int k0 = kc * 32;
            v16h a0 = load_a_frag(hh16, HID, 0,  k0, lane);
            v16h a1 = load_a_frag(hh16, HID, 16, k0, lane);
#pragma unroll
            for (int c = 0; c < 6; ++c) {
                int col0 = (wave * 6 + c) * 16;
                v16h b = load_b_frag(whh16, HID, col0, k0, lane);
                acc[0][c] = wmma16(a0, b, acc[0][c]);
                acc[1][c] = wmma16(a1, b, acc[1][c]);
            }
        }
#pragma unroll
        for (int r = 0; r < 2; ++r) {
            int n0 = r * 16 + 8 * (lane >> 4);
#pragma unroll
            for (int c = 0; c < 6; ++c) {
                int col = (wave * 6 + c) * 16 + (lane & 15);
#pragma unroll
                for (int g = 0; g < 8; ++g) stage[(size_t)(n0 + g) * G3 + col] = acc[r][c][g];
            }
        }
        __syncthreads();

        // gates + state update
        for (int idx = tid; idx < NB * HID; idx += 512) {
            int n = idx >> 9, j = idx & 511;
            size_t gb = (size_t)(t * NB + n) * G3;
            float gir = GI[gb + j], giz = GI[gb + HID + j], gin = GI[gb + 2 * HID + j];
            size_t sb = (size_t)n * G3;
            float ghr = stage[sb + j]           + b_hh[j];
            float ghz = stage[sb + HID + j]     + b_hh[HID + j];
            float ghn = stage[sb + 2 * HID + j] + b_hh[2 * HID + j];
            float rg = sigm(gir + ghr);
            float zg = sigm(giz + ghz);
            float ng = tanhf(gin + rg * ghn);
            float hn = (1.f - zg) * ng + zg * h32[idx];
            h32[idx]  = hn;
            hh16[idx] = (_Float16)hn;
            Hrelu16[(size_t)(t * NB + n) * HID + j] = (_Float16)fmaxf(hn, 0.f);
        }
        __syncthreads();

        // critic: val[n] = h . Wc + bc  (16 lanes per n)
        {
            int n = tid >> 4, l = tid & 15;
            float s = 0.f;
            for (int j = l; j < HID; j += 16) s += h32[n * HID + j] * Wc[j];
#pragma unroll
            for (int off = 8; off; off >>= 1) s += __shfl_xor(s, off, 16);
            if (l == 0) out[(size_t)(t * NB + n) * (NACT + 1) + NACT] = s + bc[0];
        }
        __syncthreads();
    }
}

// ---------------- XI = relu(H) @ Wf.T + bf : [1024,512]x[512,3616] ----------------
__global__ void k_xi_gemm(const _Float16* __restrict__ Hrelu16, const _Float16* __restrict__ Wf16,
                          const float* __restrict__ bf, float* __restrict__ XI) {
    int lane = threadIdx.x & 31, wave = threadIdx.x >> 5;
    int gid = blockIdx.x * 8 + wave;      // 14464 waves = 64 row-tiles x 226 col-tiles
    int rt = gid & 63;
    int ct = gid >> 6;
    v8f acc = zero_v8f();
    for (int kc = 0; kc < 16; ++kc) {
        v16h a = load_a_frag(Hrelu16, HID, rt * 16, kc * 32, lane);
        v16h b = load_b_frag(Wf16, HID, ct * 16, kc * 32, lane);
        acc = wmma16(a, b, acc);
    }
    int col = ct * 16 + (lane & 15);
    int r0  = rt * 16 + 8 * (lane >> 4);
    float bias = (col < XI_RAW) ? bf[col] : 0.f;
#pragma unroll
    for (int g = 0; g < 8; ++g) XI[(size_t)(r0 + g) * XI_PAD + col] = acc[g] + bias;
}

// ---------------- DNC memory recurrence: one WG per batch element ----------------
__global__ void __launch_bounds__(256)
k_mem(const float* __restrict__ XI, const float* __restrict__ Wa, const float* __restrict__ ba,
      float* __restrict__ out, float* __restrict__ Lbase) {
    extern __shared__ float Mdyn[];               // M: 128 x 512 f32 (256 KB)
    __shared__ float s_wr[HEADS][MEMN], s_wrn[HEADS][MEMN];
    __shared__ float s_u[MEMN], s_ww[MEMN], s_p[MEMN], s_alloc[MEMN];
    __shared__ float s_cw[MEMN], s_usort[MEMN], s_excl[MEMN];
    __shared__ int   s_rank[MEMN];
    __shared__ float s_kw[HID], s_e[HID], s_v[HID], s_Kr[HEADS][HID];
    __shared__ float s_fwd[HEADS][MEMN], s_bwd[HEADS][MEMN], s_bwdp[8][HEADS][MEMN];
    __shared__ float s_cosr[HEADS][MEMN], s_rread[HID], s_logits[NACT];
    __shared__ float s_scal[32]; // 0:kwn 1:bw 2:ga 3:gw 4..7:Br 8..11:F 12..15:Krn 16..27:Pi 28:sumww

    const int tid = threadIdx.x, lane = tid & 31, wave = tid >> 5;
    const int n = blockIdx.x;
    float* Lg = Lbase + (size_t)n * MEMN * MEMN;

    // init state
    for (int i = tid; i < MEMN * HID; i += 256) Mdyn[i] = 0.f;
    for (int i = tid; i < MEMN * MEMN; i += 256) Lg[i] = 0.f;
    for (int i = tid; i < HEADS * MEMN; i += 256) s_wr[i >> 7][i & 127] = 0.f;
    if (tid < MEMN) { s_u[tid] = 0.f; s_ww[tid] = 0.f; s_p[tid] = 0.f; }
    __syncthreads();

    for (int t = 0; t < T_STEPS; ++t) {
        const float* xi = XI + (size_t)(t * NB + n) * XI_PAD;

        // --- stage xi slices + scalar nonlinearities ---
        for (int i = tid; i < HEADS * HID; i += 256) s_Kr[i >> 9][i & 511] = xi[OFF_KR + i];
        for (int j = tid; j < HID; j += 256) {
            s_kw[j] = xi[OFF_KW + j];
            s_e[j]  = sigm(xi[OFF_E + j]);
            s_v[j]  = xi[OFF_V + j];
        }
        if (tid == 0) {
            s_scal[1] = sftp(xi[OFF_BW]);
            s_scal[2] = sigm(xi[OFF_GA]);
            s_scal[3] = sigm(xi[OFF_GW]);
            for (int h = 0; h < HEADS; ++h) {
                s_scal[4 + h] = sftp(xi[OFF_BR + h]);
                s_scal[8 + h] = sigm(xi[OFF_F + h]);
                float a0 = xi[OFF_PI + h * 3], a1 = xi[OFF_PI + h * 3 + 1], a2 = xi[OFF_PI + h * 3 + 2];
                float mx = fmaxf(a0, fmaxf(a1, a2));
                float e0 = expf(a0 - mx), e1 = expf(a1 - mx), e2 = expf(a2 - mx);
                float inv = 1.f / (e0 + e1 + e2);
                s_scal[16 + h * 3] = e0 * inv; s_scal[17 + h * 3] = e1 * inv; s_scal[18 + h * 3] = e2 * inv;
            }
        }
        __syncthreads();

        // --- norms of kw and Kr[h] ---
        if (wave == 0) {
            float s = 0.f;
            for (int j = lane; j < HID; j += 32) s += s_kw[j] * s_kw[j];
            s = wave_sum(s);
            if (lane == 0) s_scal[0] = sqrtf(s);
        } else if (wave <= HEADS) {
            int h = wave - 1;
            float s = 0.f;
            for (int j = lane; j < HID; j += 32) s += s_Kr[h][j] * s_Kr[h][j];
            s = wave_sum(s);
            if (lane == 0) s_scal[12 + h] = sqrtf(s);
        }
        __syncthreads();

        // --- retention psi, usage u ---
        if (tid < MEMN) {
            float psi = 1.f;
#pragma unroll
            for (int h = 0; h < HEADS; ++h) psi *= (1.f - s_scal[8 + h] * s_wr[h][tid]);
            float u = s_u[tid], w = s_ww[tid];
            s_u[tid] = (u + w - u * w) * psi;
        }
        __syncthreads();

        // --- allocation via stable rank sort + serial cumprod ---
        if (tid < MEMN) {
            float um = s_u[tid];
            int r = 0;
            for (int j = 0; j < MEMN; ++j) {
                float uj = s_u[j];
                r += (uj < um) || (uj == um && j < tid);
            }
            s_usort[r] = um;
            s_rank[tid] = r;
        }
        __syncthreads();
        if (tid == 0) {
            float run = 1.f;
            for (int k = 0; k < MEMN; ++k) { s_excl[k] = run; run *= s_usort[k]; }
        }
        __syncthreads();
        if (tid < MEMN) s_alloc[tid] = (1.f - s_u[tid]) * s_excl[s_rank[tid]];
        __syncthreads();

        // --- content write weights: cos(M, kw) ---
        for (int i = 0; i < 16; ++i) {
            int m = wave * 16 + i;
            float s2 = 0.f, sk = 0.f;
            for (int j = lane; j < HID; j += 32) {
                float mv = Mdyn[m * HID + j];
                s2 += mv * mv; sk += mv * s_kw[j];
            }
            s2 = wave_sum(s2); sk = wave_sum(sk);
            if (lane == 0) s_cw[m] = sk / (sqrtf(s2) * s_scal[0] + 1e-8f);
        }
        __syncthreads();
        if (tid == 0) {  // softmax(bw * cos)
            float bw = s_scal[1], mx = -1e30f;
            for (int m = 0; m < MEMN; ++m) mx = fmaxf(mx, bw * s_cw[m]);
            float sum = 0.f;
            for (int m = 0; m < MEMN; ++m) { float e = expf(bw * s_cw[m] - mx); s_usort[m] = e; sum += e; }
            float inv = 1.f / sum;
            for (int m = 0; m < MEMN; ++m) s_cw[m] = s_usort[m] * inv;
        }
        __syncthreads();
        if (tid < MEMN)
            s_ww[tid] = s_scal[3] * (s_scal[2] * s_alloc[tid] + (1.f - s_scal[2]) * s_cw[tid]);
        __syncthreads();
        if (tid == 0) {
            float s = 0.f;
            for (int m = 0; m < MEMN; ++m) s += s_ww[m];
            s_scal[28] = s;
        }
        __syncthreads();
        if (tid < MEMN) s_p[tid] = (1.f - s_scal[28]) * s_p[tid] + s_ww[tid];
        __syncthreads();

        // --- memory write (erase + add) ---
        for (int i4 = tid; i4 < MEMN * HID / 4; i4 += 256) {
            int m = i4 >> 7, d = (i4 << 2) & 511;
            float wwm = s_ww[m];
            float4 Mv = ((float4*)Mdyn)[i4];
            Mv.x = Mv.x * (1.f - wwm * s_e[d])     + wwm * s_v[d];
            Mv.y = Mv.y * (1.f - wwm * s_e[d + 1]) + wwm * s_v[d + 1];
            Mv.z = Mv.z * (1.f - wwm * s_e[d + 2]) + wwm * s_v[d + 2];
            Mv.w = Mv.w * (1.f - wwm * s_e[d + 3]) + wwm * s_v[d + 3];
            ((float4*)Mdyn)[i4] = Mv;
        }
        __syncthreads();

        // --- link matrix update fused with fwd dot + bwd partials ---
        float bp[HEADS][4];
#pragma unroll
        for (int h = 0; h < HEADS; ++h)
#pragma unroll
            for (int jr = 0; jr < 4; ++jr) bp[h][jr] = 0.f;
        for (int il = 0; il < 16; ++il) {
            int i = wave * 16 + il;
            float wwi = s_ww[i];
            float fa[HEADS] = {0.f, 0.f, 0.f, 0.f};
#pragma unroll
            for (int jr = 0; jr < 4; ++jr) {
                int j = lane + jr * 32;
                float Lij = (1.f - wwi - s_ww[j]) * Lg[i * MEMN + j] + wwi * s_p[j];
                Lg[i * MEMN + j] = Lij;
#pragma unroll
                for (int h = 0; h < HEADS; ++h) {
                    fa[h] += Lij * s_wr[h][j];
                    bp[h][jr] += Lij * s_wr[h][i];
                }
            }
#pragma unroll
            for (int h = 0; h < HEADS; ++h) {
                float s = wave_sum(fa[h]);
                if (lane == 0) s_fwd[h][i] = s;
            }
        }
#pragma unroll
        for (int h = 0; h < HEADS; ++h)
#pragma unroll
            for (int jr = 0; jr < 4; ++jr) s_bwdp[wave][h][lane + jr * 32] = bp[h][jr];
        __syncthreads();
        for (int idx = tid; idx < HEADS * MEMN; idx += 256) {
            int h = idx >> 7, m = idx & 127;
            float s = 0.f;
#pragma unroll
            for (int w = 0; w < 8; ++w) s += s_bwdp[w][h][m];
            s_bwd[h][m] = s;
        }
        __syncthreads();

        // --- content read weights: cos(Mnew, Kr[h]) ---
        for (int i = 0; i < 16; ++i) {
            int m = wave * 16 + i;
            float s2 = 0.f, d0 = 0.f, d1 = 0.f, d2 = 0.f, d3 = 0.f;
            for (int j = lane; j < HID; j += 32) {
                float mv = Mdyn[m * HID + j];
                s2 += mv * mv;
                d0 += mv * s_Kr[0][j]; d1 += mv * s_Kr[1][j];
                d2 += mv * s_Kr[2][j]; d3 += mv * s_Kr[3][j];
            }
            s2 = wave_sum(s2); d0 = wave_sum(d0); d1 = wave_sum(d1); d2 = wave_sum(d2); d3 = wave_sum(d3);
            if (lane == 0) {
                float Mn = sqrtf(s2);
                s_cosr[0][m] = d0 / (Mn * s_scal[12] + 1e-8f);
                s_cosr[1][m] = d1 / (Mn * s_scal[13] + 1e-8f);
                s_cosr[2][m] = d2 / (Mn * s_scal[14] + 1e-8f);
                s_cosr[3][m] = d3 / (Mn * s_scal[15] + 1e-8f);
            }
        }
        __syncthreads();
        if (tid < HEADS) {  // per-head softmax(Br * cos)
            int h = tid;
            float Br = s_scal[4 + h], mx = -1e30f;
            for (int m = 0; m < MEMN; ++m) mx = fmaxf(mx, Br * s_cosr[h][m]);
            float sum = 0.f;
            for (int m = 0; m < MEMN; ++m) { float e = expf(Br * s_cosr[h][m] - mx); s_bwdp[h][0][m] = e; sum += e; }
            float inv = 1.f / sum;
            for (int m = 0; m < MEMN; ++m) s_cosr[h][m] = s_bwdp[h][0][m] * inv;
        }
        __syncthreads();
        for (int idx = tid; idx < HEADS * MEMN; idx += 256) {
            int h = idx >> 7, m = idx & 127;
            s_wrn[h][m] = s_scal[16 + h * 3] * s_bwd[h][m]
                        + s_scal[17 + h * 3] * s_cosr[h][m]
                        + s_scal[18 + h * 3] * s_fwd[h][m];
        }
        __syncthreads();
        for (int idx = tid; idx < HEADS * MEMN; idx += 256) s_wr[idx >> 7][idx & 127] = s_wrn[idx >> 7][idx & 127];
        if (tid < MEMN) {
            float s = 0.f;
#pragma unroll
            for (int h = 0; h < HEADS; ++h) s += s_wrn[h][tid];
            s_cw[tid] = s;   // reuse as head-summed read weight
        }
        __syncthreads();

        // --- read vectors: r[d] = sum_m wsum[m] * M[m][d] ---
        for (int d = tid; d < HID; d += 256) {
            float s = 0.f;
            for (int m = 0; m < MEMN; ++m) s += s_cw[m] * Mdyn[m * HID + d];
            s_rread[d] = s;
        }
        __syncthreads();

        // --- actor head ---
        if (tid < NACT) {
            float s = ba[tid];
            for (int d = 0; d < HID; ++d) s += s_rread[d] * Wa[tid * HID + d];
            s_logits[tid] = s;
        }
        __syncthreads();
        if (tid == 0) {
            float mx = -1e30f;
            for (int k = 0; k < NACT; ++k) mx = fmaxf(mx, s_logits[k]);
            float sum = 0.f;
            for (int k = 0; k < NACT; ++k) { float e = expf(s_logits[k] - mx); s_usort[k] = e; sum += e; }
            float inv = 1.f / sum;
            size_t ob = (size_t)(t * NB + n) * (NACT + 1);
            for (int k = 0; k < NACT; ++k) out[ob + k] = s_usort[k] * inv;
        }
        __syncthreads();
    }
}

// ---------------- launcher ----------------
extern "C" void kernel_launch(void* const* d_in, const int* in_sizes, int n_in,
                              void* d_out, int out_size, void* d_ws, size_t ws_size,
                              hipStream_t stream) {
    (void)in_sizes; (void)n_in; (void)out_size; (void)ws_size;
    const float* inputs = (const float*)d_in[0];
    const float* w_ih   = (const float*)d_in[1];
    const float* w_hh   = (const float*)d_in[2];
    const float* b_ih   = (const float*)d_in[3];
    const float* b_hh   = (const float*)d_in[4];
    const float* Wf     = (const float*)d_in[5];
    const float* bf     = (const float*)d_in[6];
    const float* Wa     = (const float*)d_in[7];
    const float* ba     = (const float*)d_in[8];
    const float* Wc     = (const float*)d_in[9];
    const float* bc     = (const float*)d_in[10];
    float* out = (float*)d_out;

    uint8_t* p = (uint8_t*)d_ws;
    auto carve = [&](size_t bytes) { uint8_t* r = p; p += (bytes + 255) & ~(size_t)255; return (void*)r; };

    _Float16* x16     = (_Float16*)carve((size_t)T_STEPS * NB * DOBS * 2);
    _Float16* wih16   = (_Float16*)carve((size_t)G3 * DOBS * 2);
    _Float16* whh16   = (_Float16*)carve((size_t)G3 * HID * 2);
    _Float16* Wf16    = (_Float16*)carve((size_t)XI_PAD * HID * 2);
    float*    GI      = (float*)carve((size_t)T_STEPS * NB * G3 * 4);
    _Float16* Hrelu16 = (_Float16*)carve((size_t)T_STEPS * NB * HID * 2);
    float*    XIbuf   = (float*)carve((size_t)T_STEPS * NB * XI_PAD * 4);
    float*    Lbuf    = (float*)carve((size_t)NB * MEMN * MEMN * 4);

    const int nx = T_STEPS * NB * DOBS;
    k_cvt_f16<<<(nx + 255) / 256, 256, 0, stream>>>(inputs, x16, nx);
    k_cvt_f16<<<(G3 * DOBS + 255) / 256, 256, 0, stream>>>(w_ih, wih16, G3 * DOBS);
    k_cvt_f16<<<(G3 * HID + 255) / 256, 256, 0, stream>>>(w_hh, whh16, G3 * HID);
    k_pack_wf<<<(XI_PAD * HID + 255) / 256, 256, 0, stream>>>(Wf, Wf16);

    k_gi_gemm<<<768, 256, 0, stream>>>(x16, wih16, b_ih, GI);

    size_t gru_lds = (size_t)(NB * HID + NB * G3) * 4 + (size_t)NB * HID * 2;  // 294912 B
    k_gru<<<1, 512, gru_lds, stream>>>(GI, whh16, b_hh, Wc, bc, Hrelu16, out);

    k_xi_gemm<<<1808, 256, 0, stream>>>(Hrelu16, Wf16, bf, XIbuf);

    size_t mem_lds = (size_t)MEMN * HID * 4;  // 262144 B dynamic (M matrix LDS-resident)
    k_mem<<<NB, 256, mem_lds, stream>>>(XIbuf, Wa, ba, out, Lbuf);
}
